// DeepseekV3DecoderLayer_88630945120718
// MI455X (gfx1250) — compile-verified, matrix operand order checked by reading, not run on hardware
//
#include <hip/hip_runtime.h>
#include <math.h>

// ---------------------------------------------------------------------------
// DeepSeek-V3 decoder layer on MI455X (gfx1250, wave32, WMMA).
// Matrix-core bound (~1.25 TFLOP vs ~17us of HBM traffic at 23.3 TB/s), so all
// heavy math goes through v_wmma_f32_16x16x32_bf16 with fp32 accumulate.
// GEMM is a double-buffered software pipeline: GLOBAL_LOAD_ASYNC_TO_LDS_B128
// (ASYNCcnt) copies tile i+1 while WMMAs consume tile i from LDS.
// ---------------------------------------------------------------------------

typedef unsigned short bf16_t;                                   // raw bf16 storage
typedef __attribute__((ext_vector_type(16))) __bf16 v16bf;       // WMMA A/B frag
typedef __attribute__((ext_vector_type(8)))  float  v8f;         // WMMA C/D frag

#define B_    4
#define T_    2048
#define D_    2048
#define H_    16
#define QR_   1536
#define KVR_  512
#define NOPE_ 128
#define RD_   64
#define VD_   128
#define QKD_  192
#define DFF_  8192
#define BT_   (B_*T_)

// Flip to 0 if the async-to-LDS mnemonic is rejected by a toolchain.
#define USE_ASYNC_COPY 1

__device__ __forceinline__ bf16_t f2bf(float f) {
  union { float f; unsigned int u; } c; c.f = f;
  unsigned int u = c.u;
  u += 0x7FFFu + ((u >> 16) & 1u);     // round-to-nearest-even
  return (bf16_t)(u >> 16);
}

union FragBF { v16bf v; uint4 q[2]; };

__device__ __forceinline__ v8f wmma_bf16(const FragBF& a, const FragBF& b, v8f c) {
  // (neg_a, A, neg_b, B, c_mod, C, reuse_a, reuse_b)
  return __builtin_amdgcn_wmma_f32_16x16x32_bf16(false, a.v, false, b.v,
                                                 (short)0, c, false, false);
}

#if USE_ASYNC_COPY
// Per-lane async 16B copy global -> LDS (tracked with ASYNCcnt).
// Flat pointer to __shared__ has the LDS byte address in its low 32 bits
// (flat->LDS aperture truncation, ISA 10.2).
__device__ __forceinline__ void ldsa_b128(bf16_t* ldst, const bf16_t* gsrc) {
  unsigned loff = (unsigned)(size_t)ldst;
  asm volatile("global_load_async_to_lds_b128 %0, %1, off"
               :: "v"(loff), "v"(gsrc) : "memory");
}
__device__ __forceinline__ void wait_async() {
  asm volatile("s_wait_asynccnt 0x0" ::: "memory");
}
#else
__device__ __forceinline__ void wait_async() {}
#endif

// ---------------------------------------------------------------------------
// fp32 -> bf16 conversion (weights, intermediate activations)
// ---------------------------------------------------------------------------
__global__ __launch_bounds__(256)
void dsv3_f32_to_bf16(const float* __restrict__ in, bf16_t* __restrict__ out, size_t n) {
  size_t i = (size_t)blockIdx.x * blockDim.x + threadIdx.x;
  size_t stride = (size_t)gridDim.x * blockDim.x;
  for (; i < n; i += stride) out[i] = f2bf(in[i]);
}

// ---------------------------------------------------------------------------
// Row-wise RMSNorm -> bf16. One 256-thread block per row.
// ---------------------------------------------------------------------------
__global__ __launch_bounds__(256)
void dsv3_rmsnorm_bf16(const float* __restrict__ x, const float* __restrict__ w,
                       bf16_t* __restrict__ out, int cols, int in_stride) {
  const int row = blockIdx.x;
  const float* xr = x + (size_t)row * in_stride;
  float ss = 0.f;
  for (int i = threadIdx.x; i < cols; i += blockDim.x) { float v = xr[i]; ss += v * v; }
  #pragma unroll
  for (int m = 16; m >= 1; m >>= 1) ss += __shfl_xor(ss, m, 32);
  __shared__ float red[8];
  if ((threadIdx.x & 31) == 0) red[threadIdx.x >> 5] = ss;
  __syncthreads();
  __shared__ float s_inv;
  if (threadIdx.x < 8) {
    float tot = red[threadIdx.x];
    #pragma unroll
    for (int m = 4; m >= 1; m >>= 1) tot += __shfl_xor(tot, m, 32);
    if (threadIdx.x == 0) s_inv = rsqrtf(tot / (float)cols + 1e-6f);
  }
  __syncthreads();
  const float inv = s_inv;
  bf16_t* orow = out + (size_t)row * cols;
  for (int i = threadIdx.x; i < cols; i += blockDim.x)
    orow[i] = f2bf(xr[i] * inv * w[i]);
}

// ---------------------------------------------------------------------------
// Tiled bf16 WMMA GEMM: C[M,N] = A[M,K] x B[K,N] (+ Cin), fp32 accumulate.
// Block tile 128x128, 8 waves (4 M x 2 N), wave tile 32x64 = 2x4 WMMA tiles.
// Double-buffered pipeline:
//   A tile:  async global->LDS b128 copies for tile i+1 in flight during the
//            WMMAs of tile i (ASYNCcnt; one s_wait_asynccnt per iteration).
//   B tile:  coalesced b128 global loads issued right after the barrier
//            (latency hidden under WMMAs), b16 scatter into transposed LDS
//            [n][k] committed after the compute of the current tile.
// Fragment reads are two ds_load_b128 each (ISA 7.12.2 16-bit striping).
// Requires N % 16 == 0 (holds for all GEMMs in this layer).
// ---------------------------------------------------------------------------
union BChunk { bf16_t h[16]; uint4 q[2]; };

__global__ __launch_bounds__(256)
void dsv3_gemm_bf16(const bf16_t* __restrict__ A, const bf16_t* __restrict__ Bw,
                    const float* __restrict__ Cin, float* __restrict__ C,
                    int M, int N, int K) {
  __shared__ __align__(16) bf16_t sA[2][128 * 32];
  __shared__ __align__(16) bf16_t sB[2][128 * 32];   // [n][k]
  const int tid  = threadIdx.x;
  const int wave = tid >> 5;
  const int lane = tid & 31;
  const int l15  = lane & 15;
  const int hi   = lane >> 4;
  const int bm   = blockIdx.y * 128;
  const int bn   = blockIdx.x * 128;
  const int wm   = (wave >> 1) * 32;
  const int wn   = (wave & 1) * 64;

  v8f acc[2][4];
  #pragma unroll
  for (int i = 0; i < 2; ++i)
    #pragma unroll
    for (int j = 0; j < 4; ++j)
      #pragma unroll
      for (int r = 0; r < 8; ++r) acc[i][j][r] = 0.f;

  const int ar  = tid >> 1;            // A tile row (0..127)
  const int ac  = (tid & 1) * 16;      // A half-col base (halves)
  const int bkr = tid >> 3;            // B tile k row (0..31)
  const int bnc = (tid & 7) * 16;      // B tile n chunk base
  const int nB  = bn + bnc;            // this thread's B column chunk
  const bool bvalid = (nB < N);        // chunk fully valid (N % 16 == 0)

  // --- staging helpers -----------------------------------------------------
  auto stageA = [&](int buf, int k0) {
    const bf16_t* src = A + (size_t)(bm + ar) * K + k0 + ac;
    bf16_t* dst = &sA[buf][ar * 32 + ac];
#if USE_ASYNC_COPY
    ldsa_b128(dst, src);
    ldsa_b128(dst + 8, src + 8);
#else
    ((uint4*)dst)[0] = *(const uint4*)(src);
    ((uint4*)dst)[1] = *(const uint4*)(src + 8);
#endif
    if (k0 + 32 < K) __builtin_prefetch((const void*)(src + 32), 0, 1);
  };
  auto loadB = [&](int k0) -> BChunk {
    BChunk tb;
    if (bvalid) {
      const bf16_t* src = Bw + (size_t)(k0 + bkr) * (size_t)N + nB;
      tb.q[0] = *(const uint4*)(src);
      tb.q[1] = *(const uint4*)(src + 8);
    } else {
      #pragma unroll
      for (int j = 0; j < 16; ++j) tb.h[j] = 0;
    }
    return tb;
  };
  auto commitB = [&](int buf, const BChunk& tb) {
    #pragma unroll
    for (int j = 0; j < 16; ++j)
      sB[buf][(bnc + j) * 32 + bkr] = tb.h[j];
  };

  // --- pipelined K loop ----------------------------------------------------
  stageA(0, 0);
  BChunk tb = loadB(0);
  commitB(0, tb);

  int buf = 0;
  for (int k0 = 0; k0 < K; k0 += 32) {
    wait_async();            // tile i's async copies complete
    __syncthreads();         // ...and visible to all waves
    const bool more = (k0 + 32 < K);
    if (more) {              // tile i+1 copy in flight during tile i compute
      stageA(buf ^ 1, k0 + 32);
      tb = loadB(k0 + 32);
    }

    FragBF afrag[2], bfrag[4];
    #pragma unroll
    for (int i = 0; i < 2; ++i) {
      const int row = wm + i * 16 + l15;
      afrag[i].q[0] = *(const uint4*)(&sA[buf][row * 32 + hi * 8]);
      afrag[i].q[1] = *(const uint4*)(&sA[buf][row * 32 + 16 + hi * 8]);
    }
    #pragma unroll
    for (int j = 0; j < 4; ++j) {
      const int col = wn + j * 16 + l15;
      bfrag[j].q[0] = *(const uint4*)(&sB[buf][col * 32 + hi * 8]);
      bfrag[j].q[1] = *(const uint4*)(&sB[buf][col * 32 + 16 + hi * 8]);
    }
    #pragma unroll
    for (int i = 0; i < 2; ++i)
      #pragma unroll
      for (int j = 0; j < 4; ++j)
        acc[i][j] = wmma_bf16(afrag[i], bfrag[j], acc[i][j]);

    if (more) commitB(buf ^ 1, tb);   // ds scatter lands after the WMMA block
    buf ^= 1;
  }

  // C/D layout: VGPR r -> row (hi*8 + r), lanes (l15) -> col
  #pragma unroll
  for (int i = 0; i < 2; ++i) {
    const int rbase = bm + wm + i * 16 + hi * 8;
    #pragma unroll
    for (int j = 0; j < 4; ++j) {
      const int col = bn + wn + j * 16 + l15;
      if (col < N) {
        #pragma unroll
        for (int r = 0; r < 8; ++r) {
          const size_t idx = (size_t)(rbase + r) * (size_t)N + col;
          float v = acc[i][j][r];
          if (Cin) v += Cin[idx];
          C[idx] = v;
        }
      }
    }
  }
}

// ---------------------------------------------------------------------------
// RoPE assembly: qh bf16 [B,T,H,192] from q fp32 [B,T,H,192] (rope on last 64)
// ---------------------------------------------------------------------------
__global__ __launch_bounds__(256)
void dsv3_assemble_qh(const float* __restrict__ q, const int* __restrict__ pos,
                      bf16_t* __restrict__ qh) {
  const int bt = blockIdx.x;
  const float p = (float)pos[bt];
  const float* qr = q + (size_t)bt * (H_ * QKD_);
  bf16_t* outr = qh + (size_t)bt * (H_ * QKD_);
  for (int idx = threadIdx.x; idx < H_ * QKD_; idx += blockDim.x) {
    const int d = idx % QKD_;
    float v;
    if (d < NOPE_) v = qr[idx];
    else {
      const int i2 = d - NOPE_;
      const int i  = i2 & 31;
      const float ang = p * __powf(10000.0f, -(float)i * (1.0f / 32.0f));
      const float c = __cosf(ang), s = __sinf(ang);
      const int base = idx - d;
      const float x1 = qr[base + NOPE_ + i];
      const float x2 = qr[base + NOPE_ + 32 + i];
      v = (i2 < 32) ? (x1 * c - x2 * s) : (x2 * c + x1 * s);
    }
    outr[idx] = f2bf(v);
  }
}

// kh bf16 [B,T,H,192] (k_pass from kv, shared roped k_rot from ckv[512:576]),
// v bf16 [B,T,H,128]
__global__ __launch_bounds__(256)
void dsv3_assemble_khv(const float* __restrict__ kv, const float* __restrict__ ckv,
                       const int* __restrict__ pos,
                       bf16_t* __restrict__ kh, bf16_t* __restrict__ v) {
  const int bt = blockIdx.x;
  const float p = (float)pos[bt];
  const float* kvr = kv + (size_t)bt * (H_ * (NOPE_ + VD_));
  const float* cr  = ckv + (size_t)bt * (KVR_ + RD_) + KVR_;
  bf16_t* khr = kh + (size_t)bt * (H_ * QKD_);
  bf16_t* vr  = v  + (size_t)bt * (H_ * VD_);
  for (int idx = threadIdx.x; idx < H_ * QKD_; idx += blockDim.x) {
    const int h = idx / QKD_, d = idx % QKD_;
    float val;
    if (d < NOPE_) val = kvr[h * (NOPE_ + VD_) + d];
    else {
      const int i2 = d - NOPE_;
      const int i  = i2 & 31;
      const float ang = p * __powf(10000.0f, -(float)i * (1.0f / 32.0f));
      const float c = __cosf(ang), s = __sinf(ang);
      const float x1 = cr[i], x2 = cr[32 + i];
      val = (i2 < 32) ? (x1 * c - x2 * s) : (x2 * c + x1 * s);
    }
    khr[idx] = f2bf(val);
  }
  for (int idx = threadIdx.x; idx < H_ * VD_; idx += blockDim.x) {
    const int h = idx / VD_, d = idx % VD_;
    vr[idx] = f2bf(kvr[h * (NOPE_ + VD_) + NOPE_ + d]);
  }
}

// ---------------------------------------------------------------------------
// Flash-style MLA attention. grid = (T/64, B*H), block = 128 (4 waves).
// Each wave owns 16 query rows; K-loop streams 32-row K/V tiles via LDS.
// K tile: async global->LDS (row-major). V tile: coalesced b128 loads +
// b16 scatter into transposed [dv][t2]. Scores: 6x WMMA chunks (d=192).
// Online softmax fp32; P goes through per-wave LDS (s_wait_dscnt 0) back
// into A-fragment layout for the P x V WMMA.
// ---------------------------------------------------------------------------
__global__ __launch_bounds__(128)
void dsv3_mla_attention(const bf16_t* __restrict__ qh, const bf16_t* __restrict__ kh,
                        const bf16_t* __restrict__ vv, const int* __restrict__ amask,
                        float* __restrict__ og, float scale) {
  const int bh = blockIdx.y;
  const int b  = bh / H_;
  const int h  = bh % H_;
  const int qt = blockIdx.x;
  const int tid  = threadIdx.x;
  const int wave = tid >> 5;
  const int lane = tid & 31;
  const int l15  = lane & 15;
  const int hi   = lane >> 4;
  const int qrow0 = qt * 64 + wave * 16;

  __shared__ __align__(16) bf16_t sK[32 * QKD_];    // [t2][d]
  __shared__ __align__(16) bf16_t sV[128 * 32];     // [dv][t2]
  __shared__ __align__(16) bf16_t sP[4][16 * 32];   // per-wave P scratch

  // Q fragments for all 6 d-chunks, directly in A-frag layout from global
  FragBF qf[6];
  #pragma unroll
  for (int c = 0; c < 6; ++c) {
    const bf16_t* p = qh + ((size_t)(b * T_ + qrow0 + l15) * H_ + h) * QKD_ + c * 32 + hi * 8;
    qf[c].q[0] = *(const uint4*)(p);
    qf[c].q[1] = *(const uint4*)(p + 16);
  }

  float rmax[8], rsum[8];
  v8f oacc[8];
  #pragma unroll
  for (int r = 0; r < 8; ++r) { rmax[r] = -1e30f; rsum[r] = 0.f; }
  #pragma unroll
  for (int j = 0; j < 8; ++j)
    #pragma unroll
    for (int r = 0; r < 8; ++r) oacc[j][r] = 0.f;

  const int nkt = qt * 2 + 2;   // causal: only K tiles covering rows <= q rows
  for (int kt = 0; kt < nkt; ++kt) {
    const int krow0 = kt * 32;
    { // K tile -> LDS [32][192] row-major via async copies
      const int r  = tid >> 2;
      const int sg = (tid & 3) * 48;
      const bf16_t* src = kh + ((size_t)(b * T_ + krow0 + r) * H_ + h) * QKD_ + sg;
      bf16_t* dst = &sK[r * QKD_ + sg];
#if USE_ASYNC_COPY
      #pragma unroll
      for (int u = 0; u < 6; ++u) ldsa_b128(dst + u * 8, src + u * 8);
#else
      #pragma unroll
      for (int u = 0; u < 6; ++u) ((uint4*)dst)[u] = ((const uint4*)src)[u];
#endif
    }
    { // V tile -> LDS transposed [dv][t2]: coalesced b128 loads + b16 scatter
      const int t2  = tid >> 2;            // 0..31
      const int dvc = (tid & 3) * 32;      // dv chunk base
      union { bf16_t h[32]; uint4 q[4]; } tv;
      const bf16_t* src = vv + ((size_t)(b * T_ + krow0 + t2) * H_ + h) * VD_ + dvc;
      #pragma unroll
      for (int u = 0; u < 4; ++u) tv.q[u] = ((const uint4*)src)[u];
      #pragma unroll
      for (int j = 0; j < 32; ++j) sV[(dvc + j) * 32 + t2] = tv.h[j];
    }
    wait_async();
    __syncthreads();

    // scores S[16][32] = Q(16x192) x K^T(192x32)
    v8f s[2];
    #pragma unroll
    for (int ns = 0; ns < 2; ++ns)
      #pragma unroll
      for (int r = 0; r < 8; ++r) s[ns][r] = 0.f;
    #pragma unroll
    for (int c = 0; c < 6; ++c) {
      #pragma unroll
      for (int ns = 0; ns < 2; ++ns) {
        FragBF bf;
        const bf16_t* p = &sK[(ns * 16 + l15) * QKD_ + c * 32 + hi * 8];
        bf.q[0] = *(const uint4*)(p);
        bf.q[1] = *(const uint4*)(p + 16);
        s[ns] = wmma_bf16(qf[c], bf, s[ns]);
      }
    }

    // scale + causal/padding mask
    #pragma unroll
    for (int ns = 0; ns < 2; ++ns) {
      const int kg = krow0 + ns * 16 + l15;
      const int mk = amask[b * T_ + kg];
      #pragma unroll
      for (int r = 0; r < 8; ++r) {
        const int qg = qrow0 + hi * 8 + r;
        float val = s[ns][r] * scale;
        if (kg > qg || mk == 0) val = -1e30f;
        s[ns][r] = val;
      }
    }

    // online softmax (16-lane row reductions; wave halves independent)
    float corr[8];
    #pragma unroll
    for (int r = 0; r < 8; ++r) {
      float m2 = fmaxf(s[0][r], s[1][r]);
      #pragma unroll
      for (int m = 8; m >= 1; m >>= 1) m2 = fmaxf(m2, __shfl_xor(m2, m, 32));
      const float nm = fmaxf(rmax[r], m2);
      corr[r] = __expf(rmax[r] - nm);
      rmax[r] = nm;
      rsum[r] *= corr[r];
    }
    #pragma unroll
    for (int j = 0; j < 8; ++j)
      #pragma unroll
      for (int r = 0; r < 8; ++r) oacc[j][r] *= corr[r];

    float ps[8];
    #pragma unroll
    for (int r = 0; r < 8; ++r) ps[r] = 0.f;
    #pragma unroll
    for (int ns = 0; ns < 2; ++ns)
      #pragma unroll
      for (int r = 0; r < 8; ++r) {
        const float e = __expf(s[ns][r] - rmax[r]);
        ps[r] += e;
        sP[wave][(hi * 8 + r) * 32 + ns * 16 + l15] = f2bf(e);
      }
    #pragma unroll
    for (int r = 0; r < 8; ++r) {
      float t = ps[r];
      #pragma unroll
      for (int m = 8; m >= 1; m >>= 1) t += __shfl_xor(t, m, 32);
      rsum[r] += t;
    }

    // wave-local LDS RAW fence before re-reading P in A-frag layout
    asm volatile("s_wait_dscnt 0" ::: "memory");
    FragBF pf;
    {
      const bf16_t* pp = &sP[wave][l15 * 32 + hi * 8];
      pf.q[0] = *(const uint4*)(pp);
      pf.q[1] = *(const uint4*)(pp + 16);
    }
    #pragma unroll
    for (int j = 0; j < 8; ++j) {
      FragBF vf;
      const bf16_t* vp = &sV[(j * 16 + l15) * 32 + hi * 8];
      vf.q[0] = *(const uint4*)(vp);
      vf.q[1] = *(const uint4*)(vp + 16);
      oacc[j] = wmma_bf16(pf, vf, oacc[j]);
    }
    __syncthreads();
  }

  // normalize and store O fp32 [B,T,H,128]
  #pragma unroll
  for (int j = 0; j < 8; ++j) {
    const int col = j * 16 + l15;
    #pragma unroll
    for (int r = 0; r < 8; ++r) {
      const int row = qrow0 + hi * 8 + r;
      og[((size_t)(b * T_ + row) * H_ + h) * VD_ + col] = oacc[j][r] / rsum[r];
    }
  }
}

// ---------------------------------------------------------------------------
// SwiGLU activation: out_bf16 = silu(gate) * up
// ---------------------------------------------------------------------------
__global__ __launch_bounds__(256)
void dsv3_silu_mul(const float* __restrict__ g, const float* __restrict__ u,
                   bf16_t* __restrict__ out, size_t n) {
  size_t i = (size_t)blockIdx.x * blockDim.x + threadIdx.x;
  size_t stride = (size_t)gridDim.x * blockDim.x;
  for (; i < n; i += stride) {
    const float gv = g[i];
    const float sg = gv / (1.f + __expf(-gv));
    out[i] = f2bf(sg * u[i]);
  }
}

// ---------------------------------------------------------------------------
// Host orchestration
// ---------------------------------------------------------------------------
extern "C" void kernel_launch(void* const* d_in, const int* in_sizes, int n_in,
                              void* d_out, int out_size, void* d_ws, size_t ws_size,
                              hipStream_t stream) {
  (void)in_sizes; (void)n_in; (void)out_size; (void)ws_size;
  const float* x       = (const float*)d_in[0];
  const float* ln1_w   = (const float*)d_in[1];
  const float* wq_a    = (const float*)d_in[2];
  const float* q_a_ln  = (const float*)d_in[3];
  const float* wq_b    = (const float*)d_in[4];
  const float* wkv_a   = (const float*)d_in[5];
  const float* kv_a_ln = (const float*)d_in[6];
  const float* wkv_b   = (const float*)d_in[7];
  const float* wo      = (const float*)d_in[8];
  const float* ln2_w   = (const float*)d_in[9];
  const float* w_gate  = (const float*)d_in[10];
  const float* w_up    = (const float*)d_in[11];
  const float* w_down  = (const float*)d_in[12];
  const int*   amask   = (const int*)d_in[13];
  const int*   pos     = (const int*)d_in[14];
  float* out = (float*)d_out;

  // bump allocator over d_ws (~1.6 GB total)
  size_t off = 0;
  auto alloc = [&](size_t elems, size_t elsz) -> void* {
    off = (off + 255) & ~(size_t)255;
    void* p = (char*)d_ws + off;
    off += elems * elsz;
    return p;
  };

  // bf16 weight copies
  bf16_t* wqa_b  = (bf16_t*)alloc((size_t)D_ * QR_, 2);
  bf16_t* wqb_b  = (bf16_t*)alloc((size_t)QR_ * H_ * QKD_, 2);
  bf16_t* wkva_b = (bf16_t*)alloc((size_t)D_ * (KVR_ + RD_), 2);
  bf16_t* wkvb_b = (bf16_t*)alloc((size_t)KVR_ * H_ * (NOPE_ + VD_), 2);
  bf16_t* wo_b   = (bf16_t*)alloc((size_t)H_ * VD_ * D_, 2);
  bf16_t* wg_b   = (bf16_t*)alloc((size_t)D_ * DFF_, 2);
  bf16_t* wu_b   = (bf16_t*)alloc((size_t)D_ * DFF_, 2);
  bf16_t* wd_b   = (bf16_t*)alloc((size_t)DFF_ * D_, 2);
  // activations
  bf16_t* h_b    = (bf16_t*)alloc((size_t)BT_ * D_, 2);
  float*  t_qa   = (float*) alloc((size_t)BT_ * QR_, 4);
  bf16_t* qa_b   = (bf16_t*)alloc((size_t)BT_ * QR_, 2);
  float*  q_f32  = (float*) alloc((size_t)BT_ * H_ * QKD_, 4);
  float*  ckv    = (float*) alloc((size_t)BT_ * (KVR_ + RD_), 4);
  bf16_t* kvn_b  = (bf16_t*)alloc((size_t)BT_ * KVR_, 2);
  float*  kv_f32 = (float*) alloc((size_t)BT_ * H_ * (NOPE_ + VD_), 4);
  bf16_t* qh_b   = (bf16_t*)alloc((size_t)BT_ * H_ * QKD_, 2);
  bf16_t* kh_b   = (bf16_t*)alloc((size_t)BT_ * H_ * QKD_, 2);
  bf16_t* v_b    = (bf16_t*)alloc((size_t)BT_ * H_ * VD_, 2);
  float*  o_f32  = (float*) alloc((size_t)BT_ * H_ * VD_, 4);
  bf16_t* o_b    = (bf16_t*)alloc((size_t)BT_ * H_ * VD_, 2);
  float*  x1     = (float*) alloc((size_t)BT_ * D_, 4);
  bf16_t* h2_b   = (bf16_t*)alloc((size_t)BT_ * D_, 2);
  float*  gate   = (float*) alloc((size_t)BT_ * DFF_, 4);
  float*  up     = (float*) alloc((size_t)BT_ * DFF_, 4);
  bf16_t* act_b  = (bf16_t*)alloc((size_t)BT_ * DFF_, 2);

  auto conv = [&](const float* s, bf16_t* d, size_t n) {
    dsv3_f32_to_bf16<<<2048, 256, 0, stream>>>(s, d, n);
  };
  auto gemm = [&](const bf16_t* A, const bf16_t* Bm, const float* Cin, float* C,
                  int M, int N, int K) {
    dim3 g((N + 127) / 128, M / 128);
    dsv3_gemm_bf16<<<g, 256, 0, stream>>>(A, Bm, Cin, C, M, N, K);
  };

  // weight downcast (matrix-core bound layer -> bf16 WMMA path)
  conv(wq_a,   wqa_b,  (size_t)D_ * QR_);
  conv(wq_b,   wqb_b,  (size_t)QR_ * H_ * QKD_);
  conv(wkv_a,  wkva_b, (size_t)D_ * (KVR_ + RD_));
  conv(wkv_b,  wkvb_b, (size_t)KVR_ * H_ * (NOPE_ + VD_));
  conv(wo,     wo_b,   (size_t)H_ * VD_ * D_);
  conv(w_gate, wg_b,   (size_t)D_ * DFF_);
  conv(w_up,   wu_b,   (size_t)D_ * DFF_);
  conv(w_down, wd_b,   (size_t)DFF_ * D_);

  // --- MLA attention block ---
  dsv3_rmsnorm_bf16<<<BT_, 256, 0, stream>>>(x, ln1_w, h_b, D_, D_);
  gemm(h_b, wqa_b, nullptr, t_qa, BT_, QR_, D_);                       // h @ wq_a
  gemm(h_b, wkva_b, nullptr, ckv, BT_, KVR_ + RD_, D_);                // h @ wkv_a
  dsv3_rmsnorm_bf16<<<BT_, 256, 0, stream>>>(t_qa, q_a_ln, qa_b, QR_, QR_);
  gemm(qa_b, wqb_b, nullptr, q_f32, BT_, H_ * QKD_, QR_);              // q_b proj
  dsv3_rmsnorm_bf16<<<BT_, 256, 0, stream>>>(ckv, kv_a_ln, kvn_b, KVR_, KVR_ + RD_);
  gemm(kvn_b, wkvb_b, nullptr, kv_f32, BT_, H_ * (NOPE_ + VD_), KVR_); // kv_b proj
  dsv3_assemble_qh<<<BT_, 256, 0, stream>>>(q_f32, pos, qh_b);
  dsv3_assemble_khv<<<BT_, 256, 0, stream>>>(kv_f32, ckv, pos, kh_b, v_b);

  const float scale = 1.0f / sqrtf((float)QKD_);
  dim3 ag(T_ / 64, B_ * H_);
  dsv3_mla_attention<<<ag, 128, 0, stream>>>(qh_b, kh_b, v_b, amask, o_f32, scale);

  conv(o_f32, o_b, (size_t)BT_ * H_ * VD_);
  gemm(o_b, wo_b, x, x1, BT_, D_, H_ * VD_);                           // x + o @ wo

  // --- MLP block ---
  dsv3_rmsnorm_bf16<<<BT_, 256, 0, stream>>>(x1, ln2_w, h2_b, D_, D_);
  gemm(h2_b, wg_b, nullptr, gate, BT_, DFF_, D_);
  gemm(h2_b, wu_b, nullptr, up, BT_, DFF_, D_);
  dsv3_silu_mul<<<4096, 256, 0, stream>>>(gate, up, act_b, (size_t)BT_ * DFF_);
  gemm(act_b, wd_b, x1, out, BT_, D_, DFF_);                           // x1 + mlp
}
